// MultiAggLP_emb_27822798143888
// MI455X (gfx1250) — matrix operand order repaired and checked by reading, not compiled, not run on hardware
//
#include <hip/hip_runtime.h>
#include <math.h>

#define NEG_SLOPE 0.2f
#define NUM_COM_C 100

typedef __attribute__((ext_vector_type(16))) _Float16 v16h;
typedef __attribute__((ext_vector_type(8)))  float    v8f;

// ---------------------------------------------------------------------------
// Y[N,128] = X[N,128] @ W[128,128]  via V_WMMA_F32_16X16X32_F16 (wave32).
// One wave per 16x16 output tile; K fully unrolled (4 back-to-back WMMAs).
// VGPR packing (contiguous K order k = 16*(lane>>4) + s, same for A and B):
//   A (16x32 f16): M = lane&15; per lane the 16 halves are one contiguous
//                  K run [akb..akb+15]  -> 4x float4 loads
//   B (32x16 f16): N = lane&15; lanes<16 hold K 0..15+kb, lanes>=16 K 16..31+kb
//   C/D (16x16 f32): VGPR v -> M = v (+8 for lanes>=16), N = lane&15
// ---------------------------------------------------------------------------
__global__ void gemm128_wmma(const float* __restrict__ X, const float* __restrict__ W,
                             float* __restrict__ Y, int nrows) {
  const int lane = threadIdx.x & 31;
  const int wave = (blockIdx.x * blockDim.x + threadIdx.x) >> 5;
  const int nRowTiles = nrows >> 4;
  if (wave >= nRowTiles * 8) return;
  const int rowTile = wave >> 3;
  const int colTile = wave & 7;
  const int mrow = rowTile * 16 + (lane & 15);
  const int ncol = colTile * 16 + (lane & 15);
  const bool hi = lane >= 16;
  const float* xrow = X + (size_t)mrow * 128;
  v8f acc = {};
#pragma unroll
  for (int kb = 0; kb < 128; kb += 32) {
    const int akb = kb + (hi ? 16 : 0);
    // A: one contiguous 16-float run -> 4 x b128 loads, convert to f16
    float4 x0 = *(const float4*)(xrow + akb);
    float4 x1 = *(const float4*)(xrow + akb + 4);
    float4 x2 = *(const float4*)(xrow + akb + 8);
    float4 x3 = *(const float4*)(xrow + akb + 12);
    v16h a;
    a[0]  = (_Float16)x0.x; a[1]  = (_Float16)x0.y;
    a[2]  = (_Float16)x0.z; a[3]  = (_Float16)x0.w;
    a[4]  = (_Float16)x1.x; a[5]  = (_Float16)x1.y;
    a[6]  = (_Float16)x1.z; a[7]  = (_Float16)x1.w;
    a[8]  = (_Float16)x2.x; a[9]  = (_Float16)x2.y;
    a[10] = (_Float16)x2.z; a[11] = (_Float16)x2.w;
    a[12] = (_Float16)x3.x; a[13] = (_Float16)x3.y;
    a[14] = (_Float16)x3.z; a[15] = (_Float16)x3.w;
    // B: column gather (stride 128 floats), convert to f16
    const int bkb = kb + (hi ? 16 : 0);
    v16h b;
#pragma unroll
    for (int j = 0; j < 16; ++j)
      b[j] = (_Float16)W[(bkb + j) * 128 + ncol];
    acc = __builtin_amdgcn_wmma_f32_16x16x32_f16(false, a, false, b,
                                                 (short)0, acc, false, false);
  }
#pragma unroll
  for (int v = 0; v < 8; ++v) {
    const int r = rowTile * 16 + v + (hi ? 8 : 0);
    Y[(size_t)r * 128 + ncol] = acc[v];
  }
}

// alpha_s[n,h] = dot(h[n,h,:], a_src[h,:]);  same for alpha_d
__global__ void alphas_kernel(const float* __restrict__ h,
                              const float* __restrict__ asrc,
                              const float* __restrict__ adst,
                              float* __restrict__ alpha_s,
                              float* __restrict__ alpha_d, int n) {
  int idx = blockIdx.x * blockDim.x + threadIdx.x;
  if (idx >= n * 8) return;
  int node = idx >> 3, hh = idx & 7;
  float s = 0.f, d = 0.f;
  const float* hv = h + (size_t)node * 128 + hh * 16;
  const float* av = asrc + hh * 16;
  const float* bv = adst + hh * 16;
#pragma unroll
  for (int j = 0; j < 16; ++j) { s += hv[j] * av[j]; d += hv[j] * bv[j]; }
  alpha_s[idx] = s; alpha_d[idx] = d;
}

__global__ void fill_f32(float* __restrict__ p, float v, int n) {
  int i = blockIdx.x * blockDim.x + threadIdx.x;
  if (i < n) p[i] = v;
}

__device__ __forceinline__ void atomicMaxF(float* addr, float val) {
  if (val >= 0.f) atomicMax((int*)addr, __float_as_int(val));
  else            atomicMin((unsigned int*)addr, __float_as_uint(val));
}

__device__ __forceinline__ float leakyr(float x) {
  return x >= 0.f ? x : NEG_SLOPE * x;
}

__global__ void edge_max(const int* __restrict__ src, const int* __restrict__ dst,
                         const float* __restrict__ as, const float* __restrict__ ad,
                         float* __restrict__ mbuf, int E) {
  int e = blockIdx.x * blockDim.x + threadIdx.x;
  if (e >= E) return;
  int s = src[e], d = dst[e];
#pragma unroll
  for (int h = 0; h < 8; ++h)
    atomicMaxF(&mbuf[d * 8 + h], leakyr(as[s * 8 + h] + ad[d * 8 + h]));
}

__global__ void edge_denom(const int* __restrict__ src, const int* __restrict__ dst,
                           const float* __restrict__ ew,
                           const float* __restrict__ as, const float* __restrict__ ad,
                           const float* __restrict__ mbuf, float* __restrict__ den, int E) {
  int e = blockIdx.x * blockDim.x + threadIdx.x;
  if (e >= E) return;
  int s = src[e], d = dst[e];
  float w = ew[e];
#pragma unroll
  for (int h = 0; h < 8; ++h) {
    float ev = leakyr(as[s * 8 + h] + ad[d * 8 + h]);
    float m = mbuf[d * 8 + h];
    m = (m < -3.0e38f) ? 0.f : m;
    atomicAdd(&den[d * 8 + h], w * expf(ev - m));
  }
}

// wave-per-edge: 32 lanes x 4 floats = 128-dim message, head = lane>>2
__global__ void edge_scatter(const int* __restrict__ src, const int* __restrict__ dst,
                             const float* __restrict__ ew,
                             const float* __restrict__ as, const float* __restrict__ ad,
                             const float* __restrict__ mbuf, const float* __restrict__ den,
                             const float* __restrict__ h, float* __restrict__ out, int E) {
  int e = (blockIdx.x * blockDim.x + threadIdx.x) >> 5;
  if (e >= E) return;
  int lane = threadIdx.x & 31;
  int s = src[e], d = dst[e];
  int hh = lane >> 2;
  float ev = leakyr(as[s * 8 + hh] + ad[d * 8 + hh]);
  float m = mbuf[d * 8 + hh];
  m = (m < -3.0e38f) ? 0.f : m;
  float w = ew[e] * expf(ev - m);
  float alpha = w / (den[d * 8 + hh] + 1e-16f);
  int c0 = lane * 4;
  float4 hv = *(const float4*)(h + (size_t)s * 128 + c0);
  float* ob = out + (size_t)d * 128 + c0;
  atomicAdd(ob + 0, alpha * hv.x);
  atomicAdd(ob + 1, alpha * hv.y);
  atomicAdd(ob + 2, alpha * hv.z);
  atomicAdd(ob + 3, alpha * hv.w);
}

__global__ void elu_kernel(const float* __restrict__ in, float* __restrict__ out, int n) {
  int i = blockIdx.x * blockDim.x + threadIdx.x;
  if (i >= n) return;
  float x = in[i];
  out[i] = x > 0.f ? x : expm1f(x);
}

// pooled[part[n],c] += d_com[n]*micro[n,c]   (block per node, thread per column)
__global__ void meso_kernel(const float* __restrict__ micro, const int* __restrict__ part,
                            const float* __restrict__ dcom, float* __restrict__ pooled, int n) {
  int node = blockIdx.x;
  if (node >= n) return;
  int c = threadIdx.x;
  atomicAdd(&pooled[part[node] * 128 + c], dcom[node] * micro[(size_t)node * 128 + c]);
}

// macro_vec[c] = sum_n d_macro[n]*micro[n,c]   (thread owns a column)
__global__ void macro_kernel(const float* __restrict__ micro, const float* __restrict__ dmac,
                             float* __restrict__ macrov, int n) {
  int c = threadIdx.x;
  float acc = 0.f;
  for (int node = blockIdx.x; node < n; node += gridDim.x)
    acc += dmac[node] * micro[(size_t)node * 128 + c];
  atomicAdd(&macrov[c], acc);
}

// wave per node: parameter-free 3-way attention, write agg -> out [N,384]
__global__ void agg_attention(const float* __restrict__ micro, const float* __restrict__ pooled,
                              const float* __restrict__ macrov, const int* __restrict__ part,
                              float* __restrict__ out, int n) {
  int node = (blockIdx.x * blockDim.x + threadIdx.x) >> 5;
  if (node >= n) return;
  int lane = threadIdx.x & 31;
  int p = part[node];
  int c0 = lane * 4;
  float4 mi4 = *(const float4*)(micro + (size_t)node * 128 + c0);
  float4 me4 = *(const float4*)(pooled + p * 128 + c0);
  float4 ma4 = *(const float4*)(macrov + c0);
  float miv[4] = {mi4.x, mi4.y, mi4.z, mi4.w};
  float mev[4] = {me4.x, me4.y, me4.z, me4.w};
  float mav[4] = {ma4.x, ma4.y, ma4.z, ma4.w};
  float smm = 0.f, sme = 0.f, sma = 0.f;
#pragma unroll
  for (int j = 0; j < 4; ++j) {
    float ctx = (miv[j] + mev[j] + mav[j]) * (1.0f / 3.0f);
    smm += miv[j] * ctx; sme += mev[j] * ctx; sma += mav[j] * ctx;
  }
#pragma unroll
  for (int o = 16; o > 0; o >>= 1) {
    smm += __shfl_xor(smm, o, 32);
    sme += __shfl_xor(sme, o, 32);
    sma += __shfl_xor(sma, o, 32);
  }
  float mx = fmaxf(smm, fmaxf(sme, sma));
  float e0 = expf(smm - mx), e1 = expf(sme - mx), e2 = expf(sma - mx);
  float inv = 1.0f / (e0 + e1 + e2);
  float a0 = e0 * inv, a1 = e1 * inv, a2 = e2 * inv;
  float* ob = out + (size_t)node * 384;
#pragma unroll
  for (int j = 0; j < 4; ++j) {
    ob[c0 + j]       = a0 * miv[j];
    ob[128 + c0 + j] = a1 * mev[j];
    ob[256 + c0 + j] = a2 * mav[j];
  }
}

// per-column sum of squares over N rows (thread owns column, coalesced rows)
__global__ void colsumsq_kernel(const float* __restrict__ out, float* __restrict__ css, int n) {
  int c = threadIdx.x;          // 384 threads
  float acc = 0.f;
  for (int node = blockIdx.x; node < n; node += gridDim.x) {
    float v = out[(size_t)node * 384 + c];
    acc += v * v;
  }
  atomicAdd(&css[c], acc);
}

__global__ void colnorm_kernel(float* __restrict__ out, const float* __restrict__ css, int total) {
  int i = blockIdx.x * blockDim.x + threadIdx.x;
  if (i >= total) return;
  int c = i % 384;
  out[i] /= fmaxf(sqrtf(css[c]), 1e-12f);
}

// ---------------------------------------------------------------------------
static void run_gat_layer(const float* X, const float* W, const float* asrc, const float* adst,
                          const int* src, const int* dst, const float* ew,
                          float* hbuf, float* outbuf, float* actbuf,
                          float* as, float* ad, float* mbuf, float* den,
                          int N, int E, hipStream_t stream) {
  const int waves = (N / 16) * 8;
  gemm128_wmma<<<(waves * 32 + 127) / 128, 128, 0, stream>>>(X, W, hbuf, N);
  alphas_kernel<<<(N * 8 + 255) / 256, 256, 0, stream>>>(hbuf, asrc, adst, as, ad, N);
  fill_f32<<<(N * 8 + 255) / 256, 256, 0, stream>>>(mbuf, -__builtin_huge_valf(), N * 8);
  hipMemsetAsync(den, 0, (size_t)N * 8 * sizeof(float), stream);
  hipMemsetAsync(outbuf, 0, (size_t)N * 128 * sizeof(float), stream);
  edge_max<<<(E + 255) / 256, 256, 0, stream>>>(src, dst, as, ad, mbuf, E);
  edge_denom<<<(E + 255) / 256, 256, 0, stream>>>(src, dst, ew, as, ad, mbuf, den, E);
  edge_scatter<<<(E * 32 + 255) / 256, 256, 0, stream>>>(src, dst, ew, as, ad, mbuf, den,
                                                         hbuf, outbuf, E);
  elu_kernel<<<(N * 128 + 255) / 256, 256, 0, stream>>>(outbuf, actbuf, N * 128);
}

extern "C" void kernel_launch(void* const* d_in, const int* in_sizes, int n_in,
                              void* d_out, int out_size, void* d_ws, size_t ws_size,
                              hipStream_t stream) {
  const int*   edge_index = (const int*)d_in[0];
  const float* edge_weight= (const float*)d_in[1];
  const float* feat       = (const float*)d_in[2];
  const int*   partition  = (const int*)d_in[3];
  const float* d_com      = (const float*)d_in[4];
  const float* d_macro    = (const float*)d_in[5];
  const float* W1   = (const float*)d_in[6];
  const float* as1  = (const float*)d_in[7];
  const float* ad1  = (const float*)d_in[8];
  const float* W2   = (const float*)d_in[9];
  const float* as2  = (const float*)d_in[10];
  const float* ad2  = (const float*)d_in[11];

  const int N = in_sizes[2] / 128;
  const int E = in_sizes[1];
  const int* src = edge_index;
  const int* dst = edge_index + E;
  float* out = (float*)d_out;

  // workspace carve (256B aligned)
  char* ws = (char*)d_ws;
  size_t off = 0;
  auto carve = [&](size_t bytes) { char* p = ws + off; off += (bytes + 255) & ~(size_t)255; return p; };
  float* hbuf   = (float*)carve((size_t)N * 128 * 4);
  float* outbuf = (float*)carve((size_t)N * 128 * 4);
  float* actbuf = (float*)carve((size_t)N * 128 * 4);
  float* asb    = (float*)carve((size_t)N * 8 * 4);
  float* adb    = (float*)carve((size_t)N * 8 * 4);
  float* mbuf   = (float*)carve((size_t)N * 8 * 4);
  float* den    = (float*)carve((size_t)N * 8 * 4);
  float* pooled = (float*)carve((size_t)NUM_COM_C * 128 * 4);
  float* macrov = (float*)carve(128 * 4);
  float* css    = (float*)carve(384 * 4);
  (void)ws_size; (void)n_in; (void)out_size;

  // layer 1: feat -> actbuf ; layer 2: actbuf -> actbuf (micro)
  run_gat_layer(feat,   W1, as1, ad1, src, dst, edge_weight,
                hbuf, outbuf, actbuf, asb, adb, mbuf, den, N, E, stream);
  run_gat_layer(actbuf, W2, as2, ad2, src, dst, edge_weight,
                hbuf, outbuf, actbuf, asb, adb, mbuf, den, N, E, stream);
  float* micro = actbuf;

  // meso / macro poolings
  hipMemsetAsync(pooled, 0, (size_t)NUM_COM_C * 128 * 4, stream);
  hipMemsetAsync(macrov, 0, 128 * 4, stream);
  hipMemsetAsync(css,    0, 384 * 4, stream);
  meso_kernel<<<N, 128, 0, stream>>>(micro, partition, d_com, pooled, N);
  macro_kernel<<<256, 128, 0, stream>>>(micro, d_macro, macrov, N);

  // attention multi-aggregation -> d_out [N,384], then column-wise L2 norm
  agg_attention<<<(N * 32 + 255) / 256, 256, 0, stream>>>(micro, pooled, macrov, partition,
                                                          out, N);
  colsumsq_kernel<<<256, 384, 0, stream>>>(out, css, N);
  colnorm_kernel<<<(N * 384 + 255) / 256, 256, 0, stream>>>(out, css, N * 384);
}
